// DecoderLayer_35167192219772
// MI455X (gfx1250) — matrix operand order repaired and checked by reading, not compiled, not run on hardware
//
#include <hip/hip_runtime.h>

// ---------------------------------------------------------------------------
// MI455X (gfx1250) decoder layer: bf16 WMMA GEMMs + flash attention + LN.
// Wave32, WMMA 16x16x32 bf16 -> f32 accumulate, async global->LDS staging,
// pre-transposed weights (B direct from L2), transposed-V flash (barrier-free).
// ---------------------------------------------------------------------------

typedef __bf16 bf16;
typedef __bf16  v16bf __attribute__((ext_vector_type(16)));
typedef float   v8f   __attribute__((ext_vector_type(8)));
typedef int     v4i_g __attribute__((vector_size(4 * sizeof(int))));  // matches builtin param

union Frag {           // one WMMA 16-bf16-per-lane operand = 8 VGPRs = 2x uint4
    v16bf v;
    uint4 q[2];
};
union Pack8 {          // 8 bf16 packed for one 16B store
    uint4 q;
    unsigned short u[8];
};

__device__ __forceinline__ bf16 us2bf(unsigned short u) {
    bf16 b; __builtin_memcpy(&b, &u, 2); return b;
}
__device__ __forceinline__ unsigned short f2bf_bits(float f) {
    union { float f; unsigned u; } v; v.f = f;
    unsigned r = v.u + 0x7FFFu + ((v.u >> 16) & 1u);   // round-to-nearest-even
    return (unsigned short)(r >> 16);
}
__device__ __forceinline__ bf16 f2bf(float f) { return us2bf(f2bf_bits(f)); }

#if defined(__gfx1250__) && __has_builtin(__builtin_amdgcn_global_load_async_to_lds_b128)
#define HAVE_ASYNC_LDS 1
#else
#define HAVE_ASYNC_LDS 0
#endif

__device__ __forceinline__ void copy16_g2l(const bf16* g, bf16* l) {
#if HAVE_ASYNC_LDS
    __builtin_amdgcn_global_load_async_to_lds_b128(
        (__attribute__((address_space(1))) v4i_g*)(g),
        (__attribute__((address_space(3))) v4i_g*)(l), 0, 0);
#else
    *reinterpret_cast<uint4*>(l) = *reinterpret_cast<const uint4*>(g);
#endif
}
__device__ __forceinline__ void async_wait_all() {
#if HAVE_ASYNC_LDS
#if __has_builtin(__builtin_amdgcn_s_wait_asynccnt)
    __builtin_amdgcn_s_wait_asynccnt(0);
#else
    asm volatile("s_wait_asynccnt 0" ::: "memory");
#endif
#endif
}
__device__ __forceinline__ void wave_order() {
#if __has_builtin(__builtin_amdgcn_wave_barrier)
    __builtin_amdgcn_wave_barrier();   // keep LDS store->load order within wave
#endif
}

static constexpr int Bb = 4, Ss = 2048, Ee = 512, Hh = 8, Dd = 64, Ff = 2048;
static constexpr int BS = Bb * Ss;   // 8192 rows

// ---------------------------------------------------------------------------
// f32 -> bf16 convert (grid-stride)
// ---------------------------------------------------------------------------
__global__ void cvt_f32_bf16(const float* __restrict__ in, bf16* __restrict__ out, int n) {
    int i = blockIdx.x * blockDim.x + threadIdx.x;
    int stride = gridDim.x * blockDim.x;
    for (; i < n; i += stride) out[i] = f2bf(in[i]);
}

// ---------------------------------------------------------------------------
// Weight transpose+convert: W[K][N] f32 -> Wt[N][K] bf16 (tiled via LDS).
// Block (32,8), grid (N/32, K/32).
// ---------------------------------------------------------------------------
__global__ __launch_bounds__(256)
void cvt_transpose(const float* __restrict__ in, bf16* __restrict__ out, int K, int N) {
    __shared__ float t[32][33];
    const int n0 = blockIdx.x * 32, k0 = blockIdx.y * 32;
    const int tx = threadIdx.x, ty = threadIdx.y;
#pragma unroll
    for (int j = 0; j < 4; ++j) {
        int k = ty + j * 8;
        t[k][tx] = in[(size_t)(k0 + k) * N + n0 + tx];
    }
    __syncthreads();
#pragma unroll
    for (int j = 0; j < 4; ++j) {
        int n = ty + j * 8;
        out[(size_t)(n0 + n) * K + k0 + tx] = f2bf(t[tx][n]);
    }
}

// ---------------------------------------------------------------------------
// Tiled WMMA GEMM: C[M,N] = A[M,K](bf16) * Bt[N][K](bf16, pre-transposed)
// + bias, optional ReLU. Block tile 128x64, 8 waves, wave = 32x32.
// A: async-to-LDS double buffer.  B: direct from global/L2, reg double buffer.
// mode: 0 = f32 out, 1 = bf16 out, 2 = bf16 transposed-per-head out (for V).
// ---------------------------------------------------------------------------
__global__ __launch_bounds__(256)
void gemm_bf16(const bf16* __restrict__ A, const bf16* __restrict__ Bt,
               const float* __restrict__ bias,
               float* __restrict__ outf, bf16* __restrict__ outb,
               int M, int N, int K, int relu, int mode)
{
    __shared__ bf16 sA[2][128 * 40];   // stride 40 -> conflict-free b128 reads

    const int tid  = threadIdx.x;
    const int lane = tid & 31;
    const int wid  = tid >> 5;
    const int wm   = wid & 3;          // wave row-block (32 rows each)
    const int wn   = wid >> 2;         // wave col-block (32 cols each)
    const int hlf  = lane >> 4;
    const int l16  = lane & 15;
    const int m0   = blockIdx.y * 128;
    const int n0   = blockIdx.x * 64;

    // A staging coordinates (hoisted)
    const int rA0 = (tid * 2) >> 2,       cs0 = ((tid * 2) & 3) * 8;
    const int rA1 = (tid * 2 + 1) >> 2,   cs1 = ((tid * 2 + 1) & 3) * 8;
    const bf16* aSrc0 = A + (size_t)(m0 + rA0) * K + cs0;
    const bf16* aSrc1 = A + (size_t)(m0 + rA1) * K + cs1;

    auto stageA = [&](int buf, int koff) {
        copy16_g2l(aSrc0 + koff, &sA[buf][rA0 * 40 + cs0]);
        copy16_g2l(aSrc1 + koff, &sA[buf][rA1 * 40 + cs1]);
    };

    // B fragment base pointers: Bt row n = weight column n, k contiguous.
    const bf16* bRow[2];
#pragma unroll
    for (int cn = 0; cn < 2; ++cn)
        bRow[cn] = Bt + (size_t)(n0 + wn * 32 + cn * 16 + l16) * K + hlf * 8;
    auto loadB = [&](Frag* bf, int t) {
#pragma unroll
        for (int cn = 0; cn < 2; ++cn) {
            const bf16* p = bRow[cn] + t * 32;
            bf[cn].q[0] = *reinterpret_cast<const uint4*>(p);
            bf[cn].q[1] = *reinterpret_cast<const uint4*>(p + 16);
        }
    };

    v8f acc[2][2] = {};
    const int nkt = K >> 5;

    Frag bcur[2], bnxt[2];
    stageA(0, 0);
    loadB(bcur, 0);

    for (int t = 0; t < nkt; ++t) {
        const int cur = t & 1;
        async_wait_all();          // my async writes into sA[cur] have landed
        __syncthreads();           // everyone's tile[cur] visible; buf^1 free

        if (t + 1 < nkt) {
            stageA(cur ^ 1, (t + 1) * 32);
            loadB(bnxt, t + 1);
            __builtin_prefetch(aSrc0 + (t + 2) * 32, 0, 1);  // global_prefetch_b8
        }

        const bf16* sAc = sA[cur];
        Frag a[2];
#pragma unroll
        for (int rm = 0; rm < 2; ++rm) {
            int r  = wm * 32 + rm * 16 + l16;
            int kb = hlf * 8;      // lane<16: K{0-7,16-23}; else K{8-15,24-31}
            a[rm].q[0] = *reinterpret_cast<const uint4*>(&sAc[r * 40 + kb]);
            a[rm].q[1] = *reinterpret_cast<const uint4*>(&sAc[r * 40 + kb + 16]);
        }
#pragma unroll
        for (int rm = 0; rm < 2; ++rm)
#pragma unroll
            for (int cn = 0; cn < 2; ++cn)
                acc[rm][cn] = __builtin_amdgcn_wmma_f32_16x16x32_bf16(
                    false, a[rm].v, false, bcur[cn].v,
                    (short)0, acc[rm][cn], false, false);
        bcur[0].v = bnxt[0].v;
        bcur[1].v = bnxt[1].v;
    }

    // ---- epilogue ----
    float bv[2];
#pragma unroll
    for (int cn = 0; cn < 2; ++cn) bv[cn] = bias[n0 + wn * 32 + cn * 16 + l16];

#pragma unroll
    for (int rm = 0; rm < 2; ++rm)
#pragma unroll
        for (int cn = 0; cn < 2; ++cn) {
            const int col = n0 + wn * 32 + cn * 16 + l16;
            const int r0  = m0 + wm * 32 + rm * 16 + hlf * 8;  // 8 consecutive rows
            if (mode == 2) {
                // transposed-per-head: Vt[(b*512 + col)][s], rows contiguous in s
                Pack8 pk;
#pragma unroll
                for (int i = 0; i < 8; ++i)
                    pk.u[i] = f2bf_bits(acc[rm][cn][i] + bv[cn]);
                const int bi = r0 >> 11, s0 = r0 & (Ss - 1);
                *reinterpret_cast<uint4*>(outb + ((size_t)bi * 512 + col) * Ss + s0) = pk.q;
            } else {
#pragma unroll
                for (int i = 0; i < 8; ++i) {
                    float v = acc[rm][cn][i] + bv[cn];
                    if (relu) v = fmaxf(v, 0.0f);
                    if (mode == 1) outb[(size_t)(r0 + i) * N + col] = f2bf(v);
                    else           outf[(size_t)(r0 + i) * N + col] = v;
                }
            }
        }
}

// ---------------------------------------------------------------------------
// Flash attention (online softmax), barrier-free. 4 waves/block, wave = 16 q.
// Q,K,O: [B*S, H*64] bf16 row-major.  Vt: [(b*512+h*64+d)][s] bf16.
// K and V fragments stream directly from global/L2.
// ---------------------------------------------------------------------------
__global__ __launch_bounds__(128)
void flash_attn(const bf16* __restrict__ Q, const bf16* __restrict__ Kc,
                const bf16* __restrict__ Vt, bf16* __restrict__ O,
                const int* __restrict__ padmask, int causal, float scale)
{
    __shared__ bf16 sP[4][16 * 40];   // per-wave probability tile (C->A relayout)

    const int b    = blockIdx.z, h = blockIdx.y;
    const int lane = threadIdx.x & 31;
    const int w    = threadIdx.x >> 5;
    const int hlf  = lane >> 4;
    const int l16  = lane & 15;
    const int qt   = blockIdx.x * 64 + w * 16;
    const size_t base  = ((size_t)b * Ss) * (Hh * Dd) + (size_t)h * Dd;
    const size_t vbase = ((size_t)b * 512 + h * 64) * (size_t)Ss;

    // ---- Q fragments: A-matrix 16x64 = two 16x32 frags (kept in VGPRs) ----
    Frag qf[2];
    {
        const bf16* qrow = Q + base + (size_t)(qt + l16) * (Hh * Dd);
#pragma unroll
        for (int ds = 0; ds < 2; ++ds) {
            int c0 = ds * 32 + hlf * 8;
            qf[ds].q[0] = *reinterpret_cast<const uint4*>(qrow + c0);
            qf[ds].q[1] = *reinterpret_cast<const uint4*>(qrow + c0 + 16);
        }
    }
    // V fragment row pointers (d fixed per lane, keys contiguous)
    const bf16* vRow[4];
#pragma unroll
    for (int g = 0; g < 4; ++g)
        vRow[g] = Vt + vbase + (size_t)(g * 16 + l16) * Ss + hlf * 8;

    float mrow[8], lrow[8];
#pragma unroll
    for (int i = 0; i < 8; ++i) { mrow[i] = -1e30f; lrow[i] = 0.0f; }
    v8f o[4] = {};

    const int nkb = causal ? ((qt + 47) >> 5) : (Ss / 32);   // per-wave bound

    for (int kb = 0; kb < nkb; ++kb) {
        // ---- scores: S = Q * K^T (K row-major == B-fragment order) ----
        v8f sc[2] = {};
#pragma unroll
        for (int kg = 0; kg < 2; ++kg) {
            int key = kb * 32 + kg * 16 + l16;
            const bf16* krow = Kc + base + (size_t)key * (Hh * Dd);
#pragma unroll
            for (int ds = 0; ds < 2; ++ds) {
                Frag kf;
                int c0 = ds * 32 + hlf * 8;
                kf.q[0] = *reinterpret_cast<const uint4*>(krow + c0);
                kf.q[1] = *reinterpret_cast<const uint4*>(krow + c0 + 16);
                sc[kg] = __builtin_amdgcn_wmma_f32_16x16x32_bf16(
                    false, qf[ds].v, false, kf.v, (short)0, sc[kg], false, false);
            }
        }

        // ---- mask + online softmax; write P to LDS in A layout ----
        int k0 = kb * 32 + l16, k1 = k0 + 16;
        float pm0 = 0.0f, pm1 = 0.0f;
        if (padmask) {
            pm0 = 1e9f * (float)padmask[b * Ss + k0];
            pm1 = 1e9f * (float)padmask[b * Ss + k1];
        }
#pragma unroll
        for (int i = 0; i < 8; ++i) {
            int qr = qt + i + hlf * 8;
            float v0 = sc[0][i] * scale - pm0;
            float v1 = sc[1][i] * scale - pm1;
            if (causal) {
                if (k0 > qr) v0 = -1e9f;
                if (k1 > qr) v1 = -1e9f;
            }
            float mx = fmaxf(v0, v1);
#pragma unroll
            for (int off = 1; off < 16; off <<= 1)
                mx = fmaxf(mx, __shfl_xor(mx, off, 32));
            float mn = fmaxf(mrow[i], mx);
            float al = __expf(mrow[i] - mn);
            mrow[i] = mn;
            float e0 = __expf(v0 - mn), e1 = __expf(v1 - mn);
            float sm = e0 + e1;
#pragma unroll
            for (int off = 1; off < 16; off <<= 1)
                sm += __shfl_xor(sm, off, 32);
            lrow[i] = lrow[i] * al + sm;
#pragma unroll
            for (int g = 0; g < 4; ++g) o[g][i] = o[g][i] * al;
            int prow = i + hlf * 8;
            sP[w][prow * 40 + l16]      = f2bf(e0);
            sP[w][prow * 40 + 16 + l16] = f2bf(e1);
        }
        wave_order();   // sP is wave-private; DS in-order within wave

        // ---- O += P * V (V fragments direct from global Vt) ----
        Frag pf;
        {
            int pr = l16 * 40 + hlf * 8;
            pf.q[0] = *reinterpret_cast<const uint4*>(&sP[w][pr]);
            pf.q[1] = *reinterpret_cast<const uint4*>(&sP[w][pr + 16]);
        }
#pragma unroll
        for (int g = 0; g < 4; ++g) {
            Frag vf;
            const bf16* p = vRow[g] + kb * 32;
            vf.q[0] = *reinterpret_cast<const uint4*>(p);
            vf.q[1] = *reinterpret_cast<const uint4*>(p + 16);
            o[g] = __builtin_amdgcn_wmma_f32_16x16x32_bf16(
                false, pf.v, false, vf.v, (short)0, o[g], false, false);
        }
    }

    // ---- normalize + store ----
#pragma unroll
    for (int g = 0; g < 4; ++g)
#pragma unroll
        for (int i = 0; i < 8; ++i) {
            int r  = qt + i + hlf * 8;
            int dd = g * 16 + l16;
            O[base + (size_t)r * (Hh * Dd) + dd] = f2bf(o[g][i] / lrow[i]);
        }
}

// ---------------------------------------------------------------------------
// Residual add + LayerNorm over E=512; writes f32 result + optional bf16 copy.
// ---------------------------------------------------------------------------
__global__ __launch_bounds__(128)
void addnorm(const float* __restrict__ x, const float* __restrict__ s,
             const float* __restrict__ g, const float* __restrict__ bb,
             float* __restrict__ outf, bf16* __restrict__ outb)
{
    const int row  = blockIdx.x;
    const int tid  = threadIdx.x;
    const int lane = tid & 31, w = tid >> 5;
    const size_t off = (size_t)row * Ee + tid * 4;

    float4 xv = *reinterpret_cast<const float4*>(x + off);
    float4 sv = *reinterpret_cast<const float4*>(s + off);
    float y[4] = { xv.x + sv.x, xv.y + sv.y, xv.z + sv.z, xv.w + sv.w };
    float s1 = y[0] + y[1] + y[2] + y[3];
    float s2 = y[0]*y[0] + y[1]*y[1] + y[2]*y[2] + y[3]*y[3];
#pragma unroll
    for (int o2 = 1; o2 < 32; o2 <<= 1) {
        s1 += __shfl_xor(s1, o2, 32);
        s2 += __shfl_xor(s2, o2, 32);
    }
    __shared__ float a1[4], a2[4];
    if (lane == 0) { a1[w] = s1; a2[w] = s2; }
    __syncthreads();
    s1 = a1[0] + a1[1] + a1[2] + a1[3];
    s2 = a2[0] + a2[1] + a2[2] + a2[3];
    float mean = s1 * (1.0f / Ee);
    float var  = s2 * (1.0f / Ee) - mean * mean;
    float rstd = rsqrtf(var + 1e-3f);

    float4 ov;
    float* op = &ov.x;
#pragma unroll
    for (int j = 0; j < 4; ++j) {
        int c = tid * 4 + j;
        op[j] = (y[j] - mean) * rstd * g[c] + bb[c];
    }
    *reinterpret_cast<float4*>(outf + off) = ov;
    if (outb) {
#pragma unroll
        for (int j = 0; j < 4; ++j) outb[off + j] = f2bf(op[j]);
    }
}

// ---------------------------------------------------------------------------
// Orchestration
// ---------------------------------------------------------------------------
static void launch_cvt(const float* in, bf16* out, int n, hipStream_t st) {
    int blocks = (n + 1023) / 1024; if (blocks > 4096) blocks = 4096;
    cvt_f32_bf16<<<blocks, 256, 0, st>>>(in, out, n);
}
static void launch_cvtT(const float* in, bf16* out, int K, int N, hipStream_t st) {
    cvt_transpose<<<dim3(N / 32, K / 32), dim3(32, 8), 0, st>>>(in, out, K, N);
}

extern "C" void kernel_launch(void* const* d_in, const int* in_sizes, int n_in,
                              void* d_out, int out_size, void* d_ws, size_t ws_size,
                              hipStream_t stream) {
    (void)in_sizes; (void)n_in; (void)out_size; (void)ws_size;
    const float* x   = (const float*)d_in[0];
    const float* enc = (const float*)d_in[1];
    const int*   pad = (const int*)d_in[3];
    const float* saWq=(const float*)d_in[4],  *sabq=(const float*)d_in[5];
    const float* saWk=(const float*)d_in[6],  *sabk=(const float*)d_in[7];
    const float* saWv=(const float*)d_in[8],  *sabv=(const float*)d_in[9];
    const float* saWo=(const float*)d_in[10], *sabo=(const float*)d_in[11];
    const float* caWq=(const float*)d_in[12], *cabq=(const float*)d_in[13];
    const float* caWk=(const float*)d_in[14], *cabk=(const float*)d_in[15];
    const float* caWv=(const float*)d_in[16], *cabv=(const float*)d_in[17];
    const float* caWo=(const float*)d_in[18], *cabo=(const float*)d_in[19];
    const float* W1=(const float*)d_in[20], *b1=(const float*)d_in[21];
    const float* W2=(const float*)d_in[22], *b2=(const float*)d_in[23];
    const float* ln1g=(const float*)d_in[24], *ln1b=(const float*)d_in[25];
    const float* ln2g=(const float*)d_in[26], *ln2b=(const float*)d_in[27];
    const float* ln3g=(const float*)d_in[28], *ln3b=(const float*)d_in[29];
    float* out = (float*)d_out;

    // workspace carve-up (256B aligned)
    char* ws = (char*)d_ws; size_t off = 0;
    auto alloc = [&](size_t bytes) -> char* {
        char* p = ws + off; off += (bytes + 255) & ~(size_t)255; return p;
    };
    const size_t EBS = (size_t)BS * Ee;
    bf16* xb   = (bf16*)alloc(EBS * 2);
    bf16* eb   = (bf16*)alloc(EBS * 2);
    bf16* wbuf = (bf16*)alloc(((size_t)8 * Ee * Ee + 2 * (size_t)Ee * Ff) * 2);
    bf16* saWqT=wbuf, *saWkT=saWqT+Ee*Ee, *saWvT=saWkT+Ee*Ee, *saWoT=saWvT+Ee*Ee;
    bf16* caWqT=saWoT+Ee*Ee, *caWkT=caWqT+Ee*Ee, *caWvT=caWkT+Ee*Ee, *caWoT=caWvT+Ee*Ee;
    bf16* W1T = caWoT + Ee*Ee;              // [Ff][Ee]
    bf16* W2T = W1T + (size_t)Ee * Ff;      // [Ee][Ff]
    bf16* qb   = (bf16*)alloc(EBS * 2);
    bf16* kb_  = (bf16*)alloc(EBS * 2);
    bf16* vb   = (bf16*)alloc(EBS * 2);     // transposed-per-head Vt layout
    bf16* attn = (bf16*)alloc(EBS * 2);
    float* proj= (float*)alloc(EBS * 4);
    float* x1f = (float*)alloc(EBS * 4);
    bf16*  x1b = (bf16*)alloc(EBS * 2);
    float* x2f = (float*)alloc(EBS * 4);
    bf16*  x2b = (bf16*)alloc(EBS * 2);
    bf16*  ff1 = (bf16*)alloc((size_t)BS * Ff * 2);

    // --- converts (weights transposed to [N][K] bf16) ---
    launch_cvt(x,   xb, (int)EBS, stream);
    launch_cvt(enc, eb, (int)EBS, stream);
    launch_cvtT(saWq, saWqT, Ee, Ee, stream);
    launch_cvtT(saWk, saWkT, Ee, Ee, stream);
    launch_cvtT(saWv, saWvT, Ee, Ee, stream);
    launch_cvtT(saWo, saWoT, Ee, Ee, stream);
    launch_cvtT(caWq, caWqT, Ee, Ee, stream);
    launch_cvtT(caWk, caWkT, Ee, Ee, stream);
    launch_cvtT(caWv, caWvT, Ee, Ee, stream);
    launch_cvtT(caWo, caWoT, Ee, Ee, stream);
    launch_cvtT(W1, W1T, Ee, Ff, stream);
    launch_cvtT(W2, W2T, Ff, Ee, stream);

    dim3 gP(Ee/64, BS/128);          // 512-wide GEMMs
    dim3 gF1(Ff/64, BS/128);         // FFN up
    dim3 gAttn(Ss/64, Hh, Bb);
    const float scale = 0.125f;      // 1/sqrt(64)

    // --- self attention ---
    gemm_bf16<<<gP, 256, 0, stream>>>(xb, saWqT, sabq, nullptr, qb,  BS, Ee, Ee, 0, 1);
    gemm_bf16<<<gP, 256, 0, stream>>>(xb, saWkT, sabk, nullptr, kb_, BS, Ee, Ee, 0, 1);
    gemm_bf16<<<gP, 256, 0, stream>>>(xb, saWvT, sabv, nullptr, vb,  BS, Ee, Ee, 0, 2);
    flash_attn<<<gAttn, 128, 0, stream>>>(qb, kb_, vb, attn, nullptr, 1, scale);
    gemm_bf16<<<gP, 256, 0, stream>>>(attn, saWoT, sabo, proj, nullptr, BS, Ee, Ee, 0, 0);
    addnorm<<<BS, 128, 0, stream>>>(x, proj, ln1g, ln1b, x1f, x1b);

    // --- cross attention ---
    gemm_bf16<<<gP, 256, 0, stream>>>(x1b, caWqT, cabq, nullptr, qb,  BS, Ee, Ee, 0, 1);
    gemm_bf16<<<gP, 256, 0, stream>>>(eb,  caWkT, cabk, nullptr, kb_, BS, Ee, Ee, 0, 1);
    gemm_bf16<<<gP, 256, 0, stream>>>(eb,  caWvT, cabv, nullptr, vb,  BS, Ee, Ee, 0, 2);
    flash_attn<<<gAttn, 128, 0, stream>>>(qb, kb_, vb, attn, pad, 0, scale);
    gemm_bf16<<<gP, 256, 0, stream>>>(attn, caWoT, cabo, proj, nullptr, BS, Ee, Ee, 0, 0);
    addnorm<<<BS, 128, 0, stream>>>(x1f, proj, ln2g, ln2b, x2f, x2b);

    // --- feed forward ---
    gemm_bf16<<<gF1, 256, 0, stream>>>(x2b, W1T, b1, nullptr, ff1, BS, Ff, Ee, 1, 1);
    gemm_bf16<<<gP,  256, 0, stream>>>(ff1, W2T, b2, proj, nullptr, BS, Ee, Ff, 0, 0);
    addnorm<<<BS, 128, 0, stream>>>(x2f, proj, ln3g, ln3b, out, nullptr);
}
